// TieCommAgent_87497073754772
// MI455X (gfx1250) — matrix operand order repaired and checked by reading, not compiled, not run on hardware
//
#include <hip/hip_runtime.h>
#include <hip/hip_bf16.h>
#include <math.h>

typedef __attribute__((ext_vector_type(16))) _Float16 v16h;
typedef __attribute__((ext_vector_type(8)))  float    v8f;

#define OBS 256
#define H   128
#define NHEADS 4
#define NACT 10
#define LN_EPS 1e-5f

// ---------------------------------------------------------------------------
// utility kernels
// ---------------------------------------------------------------------------
__global__ void k_cast_f16(const float* __restrict__ s, _Float16* __restrict__ d, long n) {
  long i = (long)blockIdx.x * blockDim.x + threadIdx.x;
  if (i < n) d[i] = (_Float16)s[i];
}

__global__ void k_fill_f32(float* __restrict__ p, float v, long n) {
  long i = (long)blockIdx.x * blockDim.x + threadIdx.x;
  if (i < n) p[i] = v;
}

// Repack row-major f32 B[K,Nc] into this file's WMMA B lane layout (contiguous k per lane half).
// Per (ktile, ntile): 32 lanes x 16 halves; lane holds column (lane&15),
// K = ktile*32 + (lane>=16 ? 16 : 0) + i.
__global__ void k_repack_b(const float* __restrict__ B, _Float16* __restrict__ Bp,
                           int K, int Nc) {
  int ntiles = Nc >> 4;
  long total = (long)K * Nc;   // == (K/32)*ntiles*512
  long idx = (long)blockIdx.x * blockDim.x + threadIdx.x;
  if (idx >= total) return;
  int i    = (int)(idx & 15);
  int lane = (int)((idx >> 4) & 31);
  long rest = idx >> 9;
  int tn = (int)(rest % ntiles);
  int kt = (int)(rest / ntiles);
  int k = kt * 32 + ((lane >> 4) ? 16 : 0) + i;
  int n = tn * 16 + (lane & 15);
  Bp[idx] = (_Float16)B[(long)k * Nc + n];
}

// ---------------------------------------------------------------------------
// WMMA GEMM: C = act(A[M,K] @ B[K,Nc] + bias). One wave32 per 16x16 tile.
// A: f16 row-major (lane layout built on the fly: lane holds row (lane&15),
//    elements 0..15 -> K = k0..k0+15, k0 = kk + hi*16).
// Bp: pre-packed (contiguous 32B per lane per K-step).
// act: 0 = none, 1 = tanh, 2 = relu
// ---------------------------------------------------------------------------
__global__ void k_wmma_gemm(const _Float16* __restrict__ A, const _Float16* __restrict__ Bp,
                            const float* __restrict__ bias,
                            float* __restrict__ Cf, _Float16* __restrict__ Ch,
                            int M, int K, int Nc, int act) {
  int wave = (int)(((long)blockIdx.x * blockDim.x + threadIdx.x) >> 5);
  int lane = threadIdx.x & 31;
  int ntiles = Nc >> 4;
  int total  = (M >> 4) * ntiles;
  if (wave >= total) return;          // wave-uniform: EXEC stays all-ones
  int tn = wave % ntiles;
  int tm = wave / ntiles;
  int hi = lane >> 4;
  int lo = lane & 15;
  const _Float16* Arow = A + (long)(tm * 16 + lo) * K;
  const _Float16* Bt   = Bp + (long)tn * 512 + (long)lane * 16;
  long bstride = (long)ntiles * 512;
  v8f acc = {};
  for (int kk = 0; kk < K; kk += 32) {
    int k0 = kk + hi * 16;
    v16h a, b;
#pragma unroll
    for (int i = 0; i < 8; ++i) { a[i] = Arow[k0 + i]; a[8 + i] = Arow[k0 + 8 + i]; }
#pragma unroll
    for (int i = 0; i < 16; ++i) b[i] = Bt[i];
    __builtin_prefetch(Arow + k0 + 32, 0, 3);
    acc = __builtin_amdgcn_wmma_f32_16x16x32_f16(false, a, false, b,
                                                 (short)0, acc, false, false);
    Bt += bstride;
  }
  int col = tn * 16 + lo;
  float bv = bias ? bias[col] : 0.0f;
  int m0 = tm * 16 + hi * 8;
#pragma unroll
  for (int r = 0; r < 8; ++r) {
    float v = acc[r] + bv;
    if (act == 1) v = tanhf(v);
    else if (act == 2) v = fmaxf(v, 0.0f);
    long o = (long)(m0 + r) * Nc + col;
    if (Cf) Cf[o] = v;
    if (Ch) Ch[o] = (_Float16)v;
  }
}

// ---------------------------------------------------------------------------
// Edge phase (GATv2), memory-bound wave-per-edge kernels
// ---------------------------------------------------------------------------
__device__ __forceinline__ void atomicMaxF32(float* addr, float val) {
  unsigned int* a = (unsigned int*)addr;
  unsigned int old = __float_as_uint(*addr);
  while (__uint_as_float(old) < val) {
    unsigned int assumed = old;
    old = atomicCAS(a, assumed, __float_as_uint(val));
    if (old == assumed) break;
  }
}

__global__ void k_edge_logits(const int* __restrict__ ei, int E, int EN,
                              const float* __restrict__ xl, const float* __restrict__ xr,
                              const float* __restrict__ att, float* __restrict__ ev) {
  long gid = (long)blockIdx.x * blockDim.x + threadIdx.x;
  int e = (int)(gid >> 5);
  int lane = (int)(gid & 31);
  if (e >= EN) return;
  int s, d;
  if (e < E) { s = ei[e]; d = ei[E + e]; } else { s = e - E; d = s; }
  float4 a = ((const float4*)(xl + (long)s * H))[lane];
  float4 b = ((const float4*)(xr + (long)d * H))[lane];
  float4 w = ((const float4*)att)[lane];
  float tx = a.x + b.x, ty = a.y + b.y, tz = a.z + b.z, tw = a.w + b.w;
  tx = tx > 0.f ? tx : 0.2f * tx;
  ty = ty > 0.f ? ty : 0.2f * ty;
  tz = tz > 0.f ? tz : 0.2f * tz;
  tw = tw > 0.f ? tw : 0.2f * tw;
  float p = tx * w.x + ty * w.y + tz * w.z + tw * w.w;
#pragma unroll
  for (int off = 16; off > 0; off >>= 1) p += __shfl_xor(p, off, 32);
  if (lane == 0) ev[e] = p;
}

__global__ void k_seg_max(const int* __restrict__ ei, int E, int EN,
                          const float* __restrict__ ev, float* __restrict__ m) {
  long e = (long)blockIdx.x * blockDim.x + threadIdx.x;
  if (e >= EN) return;
  int d = (e < E) ? ei[E + e] : (int)(e - E);
  atomicMaxF32(&m[d], ev[e]);
}

__global__ void k_soft_denom(const int* __restrict__ ei, int E, int EN,
                             const float* __restrict__ ev, const float* __restrict__ m,
                             float* __restrict__ exb, float* __restrict__ den) {
  long e = (long)blockIdx.x * blockDim.x + threadIdx.x;
  if (e >= EN) return;
  int d = (e < E) ? ei[E + e] : (int)(e - E);
  float x = __expf(ev[e] - m[d]);
  exb[e] = x;
  atomicAdd(&den[d], x);
}

__global__ void k_scatter(const int* __restrict__ ei, int E, int EN,
                          const float* __restrict__ exb, const float* __restrict__ den,
                          const float* __restrict__ xl, float* __restrict__ agg) {
  long gid = (long)blockIdx.x * blockDim.x + threadIdx.x;
  int e = (int)(gid >> 5);
  int lane = (int)(gid & 31);
  if (e >= EN) return;
  int s, d;
  if (e < E) { s = ei[e]; d = ei[E + e]; } else { s = e - E; d = s; }
  float w = exb[e] / den[d];
  float4 a = ((const float4*)(xl + (long)s * H))[lane];
  float* dst = agg + (long)d * H + lane * 4;
  atomicAdd(dst + 0, w * a.x);
  atomicAdd(dst + 1, w * a.y);
  atomicAdd(dst + 2, w * a.z);
  atomicAdd(dst + 3, w * a.w);
}

__global__ void k_intra(const float* __restrict__ agg, const float* __restrict__ gb,
                        float* __restrict__ intra, _Float16* __restrict__ intra16, long n) {
  long i = (long)blockIdx.x * blockDim.x + threadIdx.x;
  if (i >= n) return;
  float v = tanhf(agg[i] + gb[i & (H - 1)]);
  intra[i] = v;
  intra16[i] = (_Float16)v;
}

__global__ void k_gather_core(const int* __restrict__ core, const float* __restrict__ intra,
                              float* __restrict__ c32, _Float16* __restrict__ c16, long n) {
  long i = (long)blockIdx.x * blockDim.x + threadIdx.x;
  if (i >= n) return;
  int g = (int)(i >> 7), col = (int)(i & (H - 1));
  float v = intra[(long)core[g] * H + col];
  c32[i] = v;
  c16[i] = (_Float16)v;
}

// ---------------------------------------------------------------------------
// fused multi-head self-attention over G core nodes (one block per head,row)
// ---------------------------------------------------------------------------
__global__ void k_attn(const float* __restrict__ qkv, _Float16* __restrict__ o16, int G) {
  __shared__ float qs[32];
  __shared__ float red[256];
  __shared__ float oacc[32];
  int tb = blockIdx.x;
  int h  = tb & (NHEADS - 1);
  int qi = tb >> 2;
  int t = threadIdx.x;
  if (t < 32) { qs[t] = qkv[(long)qi * (3 * H) + h * 32 + t]; oacc[t] = 0.f; }
  __syncthreads();
  const float scale = 0.1767766952966369f;   // 1/sqrt(32)
  float sc[4];
#pragma unroll
  for (int j = 0; j < 4; ++j) {
    int kj = t + j * 256;
    if (kj < G) {
      const float* kr = qkv + (long)kj * (3 * H) + H + h * 32;
      float p = 0.f;
#pragma unroll
      for (int dd = 0; dd < 32; ++dd) p += qs[dd] * kr[dd];
      sc[j] = p * scale;
    } else sc[j] = -3.0e38f;
  }
  float mx = fmaxf(fmaxf(sc[0], sc[1]), fmaxf(sc[2], sc[3]));
  red[t] = mx; __syncthreads();
  for (int s = 128; s > 0; s >>= 1) { if (t < s) red[t] = fmaxf(red[t], red[t + s]); __syncthreads(); }
  mx = red[0]; __syncthreads();
  float pj[4]; float sum = 0.f;
#pragma unroll
  for (int j = 0; j < 4; ++j) { pj[j] = __expf(sc[j] - mx); sum += pj[j]; }
  red[t] = sum; __syncthreads();
  for (int s = 128; s > 0; s >>= 1) { if (t < s) red[t] += red[t + s]; __syncthreads(); }
  float inv = 1.f / red[0];
  float acc[32];
#pragma unroll
  for (int dd = 0; dd < 32; ++dd) acc[dd] = 0.f;
#pragma unroll
  for (int j = 0; j < 4; ++j) {
    int kj = t + j * 256;
    if (kj < G) {
      const float* vr = qkv + (long)kj * (3 * H) + 2 * H + h * 32;
      float p = pj[j] * inv;
#pragma unroll
      for (int dd = 0; dd < 32; ++dd) acc[dd] += p * vr[dd];
    }
  }
#pragma unroll
  for (int dd = 0; dd < 32; ++dd) atomicAdd(&oacc[dd], acc[dd]);
  __syncthreads();
  if (t < 32) o16[(long)qi * H + h * 32 + t] = (_Float16)oacc[t];
}

// ---------------------------------------------------------------------------
// layernorm(resid + delta) [* optional tanh]; block per row, 128 threads
// ---------------------------------------------------------------------------
__global__ void k_ln(const float* __restrict__ resid, const float* __restrict__ delta,
                     const float* __restrict__ gain, const float* __restrict__ beta,
                     float* __restrict__ outf, _Float16* __restrict__ outh, int do_tanh) {
  __shared__ float red[H];
  int row = blockIdx.x, t = threadIdx.x;
  long o = (long)row * H + t;
  float v = resid[o] + delta[o];
  red[t] = v; __syncthreads();
  for (int s = 64; s > 0; s >>= 1) { if (t < s) red[t] += red[t + s]; __syncthreads(); }
  float mu = red[0] * (1.0f / H);
  __syncthreads();
  float dv = v - mu;
  red[t] = dv * dv; __syncthreads();
  for (int s = 64; s > 0; s >>= 1) { if (t < s) red[t] += red[t + s]; __syncthreads(); }
  float var = red[0] * (1.0f / H);
  float y = dv * rsqrtf(var + LN_EPS) * gain[t] + beta[t];
  if (do_tanh) y = tanhf(y);
  if (outf) outf[o] = y;
  if (outh) outh[o] = (_Float16)y;
}

// after = concat(group_obs[group_assign], intra) as f16 [N, 2H]
__global__ void k_after(const int* __restrict__ ga, const _Float16* __restrict__ go16,
                        const _Float16* __restrict__ intra16, _Float16* __restrict__ after16,
                        long n) {
  long i = (long)blockIdx.x * blockDim.x + threadIdx.x;
  if (i >= n) return;
  int node = (int)(i >> 8), c = (int)(i & 255);
  after16[i] = (c < H) ? go16[(long)ga[node] * H + c]
                       : intra16[(long)node * H + (c - H)];
}

// ---------------------------------------------------------------------------
// actor/value heads: wave per node; log_softmax over 10 actions + scalar value
// ---------------------------------------------------------------------------
__global__ void k_heads(const float* __restrict__ apre, const float* __restrict__ vpre,
                        const float* __restrict__ aw2, const float* __restrict__ ab2,
                        const float* __restrict__ vw2, const float* __restrict__ vb2,
                        float* __restrict__ out, int N) {
  long gid = (long)blockIdx.x * blockDim.x + threadIdx.x;
  int node = (int)(gid >> 5);
  int lane = (int)(gid & 31);
  if (node >= N) return;
  float4 a = ((const float4*)(apre + (long)node * H))[lane];
  int d0 = lane * 4;
  float lg[NACT];
#pragma unroll
  for (int j = 0; j < NACT; ++j) {
    float p = a.x * aw2[(d0 + 0) * NACT + j] + a.y * aw2[(d0 + 1) * NACT + j]
            + a.z * aw2[(d0 + 2) * NACT + j] + a.w * aw2[(d0 + 3) * NACT + j];
#pragma unroll
    for (int off = 16; off > 0; off >>= 1) p += __shfl_xor(p, off, 32);
    lg[j] = p + ab2[j];
  }
  float mx = lg[0];
#pragma unroll
  for (int j = 1; j < NACT; ++j) mx = fmaxf(mx, lg[j]);
  float s = 0.f;
#pragma unroll
  for (int j = 0; j < NACT; ++j) s += __expf(lg[j] - mx);
  float lse = __logf(s);
  if (lane == 0) {
#pragma unroll
    for (int j = 0; j < NACT; ++j) out[(long)node * NACT + j] = lg[j] - mx - lse;
  }
  float4 v = ((const float4*)(vpre + (long)node * H))[lane];
  float pv = v.x * vw2[d0 + 0] + v.y * vw2[d0 + 1] + v.z * vw2[d0 + 2] + v.w * vw2[d0 + 3];
#pragma unroll
  for (int off = 16; off > 0; off >>= 1) pv += __shfl_xor(pv, off, 32);
  if (lane == 0) out[(long)N * NACT + node] = pv + vb2[0];
}

// ---------------------------------------------------------------------------
// host side
// ---------------------------------------------------------------------------
extern "C" void kernel_launch(void* const* d_in, const int* in_sizes, int n_in,
                              void* d_out, int out_size, void* d_ws, size_t ws_size,
                              hipStream_t stream) {
  (void)n_in; (void)out_size; (void)ws_size;
  const float* local_obs   = (const float*)d_in[0];
  const int*   edge_index  = (const int*)d_in[1];
  const int*   group_assign= (const int*)d_in[2];
  const int*   core_node   = (const int*)d_in[3];
  const float* emb_w = (const float*)d_in[4];  const float* emb_b = (const float*)d_in[5];
  const float* gat_wl = (const float*)d_in[6]; const float* gat_wr = (const float*)d_in[7];
  const float* gat_att = (const float*)d_in[8]; const float* gat_bias = (const float*)d_in[9];
  const float* attn_in_w = (const float*)d_in[10]; const float* attn_in_b = (const float*)d_in[11];
  const float* attn_out_w = (const float*)d_in[12]; const float* attn_out_b = (const float*)d_in[13];
  const float* ln1_g = (const float*)d_in[14]; const float* ln1_b = (const float*)d_in[15];
  const float* ln2_g = (const float*)d_in[16]; const float* ln2_b = (const float*)d_in[17];
  const float* ffn_w1 = (const float*)d_in[18]; const float* ffn_b1 = (const float*)d_in[19];
  const float* ffn_w2 = (const float*)d_in[20]; const float* ffn_b2 = (const float*)d_in[21];
  const float* actor_w1 = (const float*)d_in[22]; const float* actor_b1 = (const float*)d_in[23];
  const float* actor_w2 = (const float*)d_in[24]; const float* actor_b2 = (const float*)d_in[25];
  const float* value_w1 = (const float*)d_in[26]; const float* value_b1 = (const float*)d_in[27];
  const float* value_w2 = (const float*)d_in[28]; const float* value_b2 = (const float*)d_in[29];

  const int N  = in_sizes[0] / OBS;
  const int E  = in_sizes[1] / 2;
  const int G  = in_sizes[3];
  const int EN = E + N;

  char* ws = (char*)d_ws;
  size_t off = 0;
  auto alloc = [&](size_t bytes) { size_t o = off; off += (bytes + 255) & ~(size_t)255; return o; };

  size_t o_obs16  = alloc((size_t)N * OBS * 2);
  size_t o_x16    = alloc((size_t)N * H * 2);
  size_t o_xl     = alloc((size_t)N * H * 4);
  size_t o_xr     = alloc((size_t)N * H * 4);
  size_t o_ev     = alloc((size_t)EN * 4);
  size_t o_exb    = alloc((size_t)EN * 4);
  size_t o_m      = alloc((size_t)N * 4);
  size_t o_den    = alloc((size_t)N * 4);
  size_t o_agg    = o_obs16;                   // reuse: obs16 dead after GEMM1
  size_t o_intra  = alloc((size_t)N * H * 4);
  size_t o_int16  = alloc((size_t)N * H * 2);
  size_t o_c32    = alloc((size_t)G * H * 4);
  size_t o_c16    = alloc((size_t)G * H * 2);
  size_t o_qkv    = alloc((size_t)G * 3 * H * 4);
  size_t o_o16    = alloc((size_t)G * H * 2);
  size_t o_atmp   = alloc((size_t)G * H * 4);
  size_t o_h132   = alloc((size_t)G * H * 4);
  size_t o_h116   = alloc((size_t)G * H * 2);
  size_t o_f116   = alloc((size_t)G * H * 2);
  size_t o_f232   = alloc((size_t)G * H * 4);
  size_t o_go16   = alloc((size_t)G * H * 2);
  size_t o_after  = alloc((size_t)N * 2 * H * 2);
  size_t o_apre   = o_xl;                      // reuse: xl dead after scatter
  size_t o_vpre   = o_xr;                      // reuse: xr dead after scatter
  size_t o_embp   = alloc((size_t)OBS * H * 2);
  size_t o_wlp    = alloc((size_t)H * H * 2);
  size_t o_wrp    = alloc((size_t)H * H * 2);
  size_t o_inp    = alloc((size_t)H * 3 * H * 2);
  size_t o_outp   = alloc((size_t)H * H * 2);
  size_t o_f1p    = alloc((size_t)H * H * 2);
  size_t o_f2p    = alloc((size_t)H * H * 2);
  size_t o_aw1p   = alloc((size_t)2 * H * H * 2);
  size_t o_vw1p   = alloc((size_t)2 * H * H * 2);

  const int TB = 256;
  auto gs = [&](long n) { return (int)((n + TB - 1) / TB); };
  auto F16 = [&](size_t o) { return (_Float16*)(ws + o); };
  auto F32 = [&](size_t o) { return (float*)(ws + o); };

  auto gemm = [&](const _Float16* A, size_t bp, const float* bias,
                  float* Cf, _Float16* Ch, int M, int K, int Nc, int act) {
    long waves = (long)(M / 16) * (Nc / 16);
    k_wmma_gemm<<<gs(waves * 32), TB, 0, stream>>>(A, F16(bp), bias, Cf, Ch, M, K, Nc, act);
  };
  auto pack = [&](const float* B, size_t bp, int K, int Nc) {
    k_repack_b<<<gs((long)K * Nc), TB, 0, stream>>>(B, F16(bp), K, Nc);
  };

  // 0) casts + weight packs
  k_cast_f16<<<gs((long)N * OBS), TB, 0, stream>>>(local_obs, F16(o_obs16), (long)N * OBS);
  pack(emb_w, o_embp, OBS, H);
  pack(gat_wl, o_wlp, H, H);
  pack(gat_wr, o_wrp, H, H);
  pack(attn_in_w, o_inp, H, 3 * H);
  pack(attn_out_w, o_outp, H, H);
  pack(ffn_w1, o_f1p, H, H);
  pack(ffn_w2, o_f2p, H, H);
  pack(actor_w1, o_aw1p, 2 * H, H);
  pack(value_w1, o_vw1p, 2 * H, H);

  // 1) x = tanh(obs @ emb_w + emb_b)   (f16 output only; feeds GEMMs)
  gemm(F16(o_obs16), o_embp, emb_b, nullptr, F16(o_x16), N, OBS, H, 1);

  // 2) GAT lifts
  gemm(F16(o_x16), o_wlp, nullptr, F32(o_xl), nullptr, N, H, H, 0);
  gemm(F16(o_x16), o_wrp, nullptr, F32(o_xr), nullptr, N, H, H, 0);

  // 3) edge softmax-aggregation
  k_fill_f32<<<gs(N), TB, 0, stream>>>(F32(o_m), -3.0e38f, N);
  k_fill_f32<<<gs(N), TB, 0, stream>>>(F32(o_den), 0.f, N);
  k_fill_f32<<<gs((long)N * H), TB, 0, stream>>>(F32(o_agg), 0.f, (long)N * H);
  k_edge_logits<<<gs((long)EN * 32), TB, 0, stream>>>(edge_index, E, EN, F32(o_xl), F32(o_xr), gat_att, F32(o_ev));
  k_seg_max<<<gs(EN), TB, 0, stream>>>(edge_index, E, EN, F32(o_ev), F32(o_m));
  k_soft_denom<<<gs(EN), TB, 0, stream>>>(edge_index, E, EN, F32(o_ev), F32(o_m), F32(o_exb), F32(o_den));
  k_scatter<<<gs((long)EN * 32), TB, 0, stream>>>(edge_index, E, EN, F32(o_exb), F32(o_den), F32(o_xl), F32(o_agg));
  k_intra<<<gs((long)N * H), TB, 0, stream>>>(F32(o_agg), gat_bias, F32(o_intra), F16(o_int16), (long)N * H);

  // 4) transformer over core nodes
  k_gather_core<<<gs((long)G * H), TB, 0, stream>>>(core_node, F32(o_intra), F32(o_c32), F16(o_c16), (long)G * H);
  gemm(F16(o_c16), o_inp, attn_in_b, F32(o_qkv), nullptr, G, H, 3 * H, 0);
  k_attn<<<G * NHEADS, 256, 0, stream>>>(F32(o_qkv), F16(o_o16), G);
  gemm(F16(o_o16), o_outp, attn_out_b, F32(o_atmp), nullptr, G, H, H, 0);
  k_ln<<<G, H, 0, stream>>>(F32(o_c32), F32(o_atmp), ln1_g, ln1_b, F32(o_h132), F16(o_h116), 0);
  gemm(F16(o_h116), o_f1p, ffn_b1, nullptr, F16(o_f116), G, H, H, 2);
  gemm(F16(o_f116), o_f2p, ffn_b2, F32(o_f232), nullptr, G, H, H, 0);
  k_ln<<<G, H, 0, stream>>>(F32(o_h132), F32(o_f232), ln2_g, ln2_b, nullptr, F16(o_go16), 1);

  // 5) broadcast + heads
  k_after<<<gs((long)N * 2 * H), TB, 0, stream>>>(group_assign, F16(o_go16), F16(o_int16), F16(o_after), (long)N * 2 * H);
  gemm(F16(o_after), o_aw1p, actor_b1, F32(o_apre), nullptr, N, 2 * H, H, 1);
  gemm(F16(o_after), o_vw1p, value_b1, F32(o_vpre), nullptr, N, 2 * H, H, 1);
  k_heads<<<gs((long)N * 32), TB, 0, stream>>>(F32(o_apre), F32(o_vpre), actor_w2, actor_b2,
                                               value_w2, value_b2, (float*)d_out, N);
}
